// CrossGRU_48241072669257
// MI455X (gfx1250) — compile-verified, matrix operand order checked by reading, not run on hardware
//
#include <hip/hip_runtime.h>
#include <hip/hip_bf16.h>
#include <math.h>

typedef __attribute__((ext_vector_type(16))) _Float16 v16h;
typedef __attribute__((ext_vector_type(8)))  float    v8f;
typedef __attribute__((ext_vector_type(8)))  unsigned v8u;

#define NN   2048
#define TT   128
#define INW  64
#define HH   256
#define G3H  768   // 3*H
#define CTN  48    // 768/16 column tiles

// ---------------------------------------------------------------------------
// Weight packing: W (768 x K) row-major f32  ->  B-fragment layout f16:
// out[ct][kc][lane][i] = W[ct*16 + (lane&15)][kc*32 + (i&7) + ((i>>3)<<4) + ((lane>>4)<<3)]
// so each WMMA B operand is 32 contiguous bytes per lane.
// ---------------------------------------------------------------------------
__global__ void pack_B_k(const float* __restrict__ W, _Float16* __restrict__ out,
                         int Kdim, int KC) {
    int idx = blockIdx.x * blockDim.x + threadIdx.x;
    int total = CTN * KC * 32 * 16;
    if (idx >= total) return;
    int i    = idx & 15;
    int lane = (idx >> 4) & 31;
    int rem  = idx >> 9;
    int kc   = rem % KC;
    int ct   = rem / KC;
    int ncol = ct * 16 + (lane & 15);
    int k    = kc * 32 + (i & 7) + ((i >> 3) << 4) + ((lane >> 4) << 3);
    out[idx] = (_Float16)W[(size_t)ncol * Kdim + k];
}

// Hidden state lives in LDS in A-fragment layout:
//   H[(kc*32 + lane)*16 + i]  with  k = kc*32 + (i&7) + ((i>>3)<<4) + ((lane>>4)<<3),
//   row m = lane & 15.
// Inverse map for the elementwise gate phase (given row, channel c):
__device__ inline int hfrag_idx(int row, int c) {
    int kc   = c >> 5;
    int kk   = c & 31;
    int lane = row + (((kk >> 3) & 1) << 4);
    int i    = (kk & 7) + ((kk >> 4) << 3);
    return (kc * 32 + lane) * 16 + i;
}

// ---------------------------------------------------------------------------
// GEMM helpers (one 16-row tile per block; wave owns 6 col-tiles ct = wave+8j)
// ---------------------------------------------------------------------------
__device__ inline void zero_acc(v8f* acc) {
#pragma unroll
    for (int j = 0; j < 6; ++j)
#pragma unroll
        for (int r = 0; r < 8; ++r) acc[j][r] = 0.0f;
}

// A from LDS hidden-state buffer in fragment layout (one 32B vector load)
template <int KC>
__device__ inline void gemm_ldsA(const _Float16* __restrict__ Hbuf,
                                 const _Float16* __restrict__ Wp,
                                 int wave, int lane, v8f* acc) {
#pragma unroll 1
    for (int kc = 0; kc < KC; ++kc) {
        const v8u* ap = (const v8u*)(Hbuf + ((size_t)(kc * 32 + lane) * 16));
        v16h a = __builtin_bit_cast(v16h, *ap);
        v16h b[6];
#pragma unroll
        for (int j = 0; j < 6; ++j) {
            const v8u* bp =
                (const v8u*)(Wp + ((size_t)((wave + 8 * j) * KC + kc) * 32 + lane) * 16);
            b[j] = __builtin_bit_cast(v16h, *bp);
        }
#pragma unroll
        for (int j = 0; j < 6; ++j)
            acc[j] = __builtin_amdgcn_wmma_f32_16x16x32_f16(
                false, a, false, b[j], (short)0, acc[j], false, false);
    }
}

// A from global x (f32), K = 64 (2 k-chunks), per timestep
__device__ inline void gemm_xA(const float* __restrict__ x, int rowbase, int t,
                               const _Float16* __restrict__ Wp,
                               int wave, int lane, v8f* acc) {
    const int kq   = (lane >> 4) << 3;
    const int mrow = lane & 15;
    const float* xrow = x + (size_t)(rowbase + mrow) * (TT * INW) + (size_t)t * INW + kq;
#pragma unroll 1
    for (int kc = 0; kc < 2; ++kc) {
        v16h a;
#pragma unroll
        for (int p = 0; p < 8; ++p) {
            int ko = kc * 32 + ((p & 3) << 1) + ((p >> 2) << 4);
            a[2 * p]     = (_Float16)xrow[ko];
            a[2 * p + 1] = (_Float16)xrow[ko + 1];
        }
        v16h b[6];
#pragma unroll
        for (int j = 0; j < 6; ++j) {
            const v8u* bp =
                (const v8u*)(Wp + ((size_t)((wave + 8 * j) * 2 + kc) * 32 + lane) * 16);
            b[j] = __builtin_bit_cast(v16h, *bp);
        }
#pragma unroll
        for (int j = 0; j < 6; ++j)
            acc[j] = __builtin_amdgcn_wmma_f32_16x16x32_f16(
                false, a, false, b[j], (short)0, acc[j], false, false);
    }
}

// C/D layout store: VGPR r -> row r + 8*(lane>>4), col = ct*16 + (lane&15)
__device__ inline void store_acc(float* __restrict__ Gdst, int wave, int lane,
                                 const v8f* acc) {
    const int mh = (lane >> 4) << 3;
    const int nc = lane & 15;
#pragma unroll
    for (int j = 0; j < 6; ++j) {
        int ct = wave + 8 * j;
#pragma unroll
        for (int r = 0; r < 8; ++r)
            Gdst[(mh + r) * G3H + ct * 16 + nc] = acc[j][r];
    }
}

__device__ inline void gru_gates(const float* __restrict__ GX, const float* __restrict__ GH,
                                 const float* __restrict__ BI, const float* __restrict__ BH,
                                 _Float16* __restrict__ Hbuf, float* __restrict__ Sout,
                                 int tid, int rowbase) {
#pragma unroll 4
    for (int s = 0; s < 16; ++s) {
        int u   = tid + 256 * s;
        int row = u >> 8;
        int c   = u & 255;
        float xr = GX[row * G3H + c]       + BI[c];
        float xz = GX[row * G3H + 256 + c] + BI[256 + c];
        float xn = GX[row * G3H + 512 + c] + BI[512 + c];
        float hr = GH[row * G3H + c]       + BH[c];
        float hz = GH[row * G3H + 256 + c] + BH[256 + c];
        float hn = GH[row * G3H + 512 + c] + BH[512 + c];
        float r  = 1.0f / (1.0f + __expf(-(xr + hr)));
        float z  = 1.0f / (1.0f + __expf(-(xz + hz)));
        float nv = tanhf(xn + r * hn);
        int hi   = hfrag_idx(row, c);
        float hprev = (float)Hbuf[hi];
        float hnew  = (1.0f - z) * nv + z * hprev;
        Hbuf[hi] = (_Float16)hnew;
        if (Sout) Sout[(size_t)(rowbase + row) * HH + c] = hnew;
    }
}

// ---------------------------------------------------------------------------
// Fused 2-layer GRU scan. 128 blocks x 16 batch rows, 8 waves/block.
// ---------------------------------------------------------------------------
__global__ __launch_bounds__(256) void gru2_k(
    const float* __restrict__ x,
    const _Float16* __restrict__ wih0p, const _Float16* __restrict__ whh0p,
    const _Float16* __restrict__ wih1p, const _Float16* __restrict__ whh1p,
    const float* __restrict__ bih0, const float* __restrict__ bhh0,
    const float* __restrict__ bih1, const float* __restrict__ bhh1,
    float* __restrict__ S) {
    extern __shared__ char smem[];
    float*    GX  = (float*)smem;                    // 16*768 f32
    float*    GH  = GX + 16 * G3H;                   // 16*768 f32
    _Float16* H0  = (_Float16*)(GH + 16 * G3H);      // 16*256 f16 (frag layout)
    _Float16* H1  = H0 + 16 * HH;                    // 16*256 f16 (frag layout)
    float*    BI0 = (float*)(H1 + 16 * HH);          // 4 x 768 f32
    float*    BH0 = BI0 + G3H;
    float*    BI1 = BH0 + G3H;
    float*    BH1 = BI1 + G3H;

    const int tid  = threadIdx.x;
    const int wave = tid >> 5;
    const int lane = tid & 31;
    const int rowbase = blockIdx.x * 16;

    for (int u = tid; u < G3H; u += 256) {
        BI0[u] = bih0[u]; BH0[u] = bhh0[u];
        BI1[u] = bih1[u]; BH1[u] = bhh1[u];
    }
    for (int u = tid; u < 16 * HH; u += 256) {
        H0[u] = (_Float16)0.0f; H1[u] = (_Float16)0.0f;
    }
    __syncthreads();

    v8f acc[6];
    for (int t = 0; t < TT; ++t) {
        // ---- layer 0 ----
        zero_acc(acc);
        gemm_xA(x, rowbase, t, wih0p, wave, lane, acc);   // GX = x_t @ Wih0^T
        store_acc(GX, wave, lane, acc);
        zero_acc(acc);
        gemm_ldsA<8>(H0, whh0p, wave, lane, acc);         // GH = h0 @ Whh0^T
        store_acc(GH, wave, lane, acc);
        __syncthreads();
        gru_gates(GX, GH, BI0, BH0, H0, nullptr, tid, rowbase);
        __syncthreads();
        // ---- layer 1 ----
        zero_acc(acc);
        gemm_ldsA<8>(H0, wih1p, wave, lane, acc);         // GX = h0_new @ Wih1^T
        store_acc(GX, wave, lane, acc);
        zero_acc(acc);
        gemm_ldsA<8>(H1, whh1p, wave, lane, acc);         // GH = h1 @ Whh1^T
        store_acc(GH, wave, lane, acc);
        __syncthreads();
        gru_gates(GX, GH, BI1, BH1, H1, (t == TT - 1) ? S : nullptr, tid, rowbase);
        __syncthreads();
    }
}

// ---------------------------------------------------------------------------
// Epilogue (tiny FLOP count -> plain f32 VALU, parallel over rows)
// ---------------------------------------------------------------------------
__device__ inline float block_reduce_sum(float v, float* red, int tid) {
    red[tid] = v; __syncthreads();
    for (int s = 128; s > 0; s >>= 1) {
        if (tid < s) red[tid] += red[tid + s];
        __syncthreads();
    }
    float r = red[0]; __syncthreads();
    return r;
}

// out[row, h*128+d] = b[h,d] + sum_i X[row,i] * W[h,i,d]   (grid = #rows, 256 thr)
__global__ __launch_bounds__(256) void proj_k(const float* __restrict__ X,
                                              const float* __restrict__ W,
                                              const float* __restrict__ bvec,
                                              float* __restrict__ Out) {
    __shared__ float Xl[256];
    int tid = threadIdx.x, rowid = blockIdx.x;
    Xl[tid] = X[(size_t)rowid * 256 + tid];
    __syncthreads();
    int h = tid >> 7, d = tid & 127;
    float acc = bvec[h * 128 + d];
    const float* Wp = W + (size_t)h * 256 * 128 + d;
    for (int i = 0; i < 256; ++i) acc += Xl[i] * Wp[i * 128];
    Out[(size_t)rowid * 256 + tid] = acc;
}

// B = softmax(qh_R . kh / sqrt(128)) @ vh  over all 2048 keys. grid = 60 (q,h)
__global__ __launch_bounds__(256) void attnB_k(const float* __restrict__ QHR,
                                               const float* __restrict__ KH,
                                               const float* __restrict__ VH,
                                               float* __restrict__ Bout) {
    __shared__ float sc[NN];
    __shared__ float qv[128];
    __shared__ float red[256];
    int tid = threadIdx.x;
    int q = blockIdx.x >> 1, h = blockIdx.x & 1;
    if (tid < 128) qv[tid] = QHR[q * 256 + h * 128 + tid];
    __syncthreads();
    const float scale = 0.08838834764831845f; // 1/sqrt(128)
    for (int n = tid; n < NN; n += 256) {
        const float* kp = KH + (size_t)n * 256 + h * 128;
        float s = 0.f;
        for (int d2 = 0; d2 < 128; ++d2) s += qv[d2] * kp[d2];
        sc[n] = s * scale;
    }
    __syncthreads();
    float m = -3.4e38f;
    for (int n = tid; n < NN; n += 256) m = fmaxf(m, sc[n]);
    red[tid] = m; __syncthreads();
    for (int s2 = 128; s2 > 0; s2 >>= 1) {
        if (tid < s2) red[tid] = fmaxf(red[tid], red[tid + s2]);
        __syncthreads();
    }
    m = red[0]; __syncthreads();
    float loc = 0.f;
    for (int n = tid; n < NN; n += 256) {
        float e = __expf(sc[n] - m);
        sc[n] = e; loc += e;
    }
    float sum = block_reduce_sum(loc, red, tid);
    if (tid < 128) {
        float o = 0.f;
        for (int n = 0; n < NN; ++n) o += sc[n] * VH[(size_t)n * 256 + h * 128 + tid];
        Bout[q * 256 + h * 128 + tid] = o / sum;
    }
}

// Per row of S: second MHA (30 keys), gate, MLP, output head. grid = 2048.
__global__ __launch_bounds__(256) void rowfinal_k(
    const float* __restrict__ S, const float* __restrict__ Wq, const float* __restrict__ bq,
    const float* __restrict__ KHB, const float* __restrict__ VHB,
    const float* __restrict__ Wg, const float* __restrict__ bg,
    const float* __restrict__ W1, const float* __restrict__ b1,
    const float* __restrict__ W2, const float* __restrict__ b2,
    float* __restrict__ Y) {
    __shared__ float Sl[256], qv[256], sp[256], smix[256], sm2[256], red[256], sc[64];
    int tid = threadIdx.x, n = blockIdx.x;
    Sl[tid] = S[(size_t)n * 256 + tid];
    __syncthreads();
    {   // qh_S
        int h = tid >> 7, d = tid & 127;
        float acc = bq[h * 128 + d];
        const float* Wp = Wq + (size_t)h * 256 * 128 + d;
        for (int i = 0; i < 256; ++i) acc += Sl[i] * Wp[i * 128];
        qv[tid] = acc;
    }
    __syncthreads();
    if (tid < 60) {
        int h = tid / 30, kk = tid % 30;
        const float* kp = KHB + kk * 256 + h * 128;
        const float* qp = qv + h * 128;
        float s = 0.f;
        for (int d2 = 0; d2 < 128; ++d2) s += qp[d2] * kp[d2];
        sc[tid] = s * 0.08838834764831845f;
    }
    __syncthreads();
    if (tid < 2) {  // per-head softmax over 30 keys
        float m = -3.4e38f;
        for (int kk = 0; kk < 30; ++kk) m = fmaxf(m, sc[tid * 30 + kk]);
        float sum = 0.f;
        for (int kk = 0; kk < 30; ++kk) {
            float e = __expf(sc[tid * 30 + kk] - m);
            sc[tid * 30 + kk] = e; sum += e;
        }
        float inv = 1.f / sum;
        for (int kk = 0; kk < 30; ++kk) sc[tid * 30 + kk] *= inv;
    }
    __syncthreads();
    {   // S' = p @ vh_B
        int h = tid >> 7, d = tid & 127;
        float o = 0.f;
        for (int kk = 0; kk < 30; ++kk) o += sc[h * 30 + kk] * VHB[kk * 256 + h * 128 + d];
        sp[tid] = o;
    }
    float ga = block_reduce_sum(Sl[tid] * Wg[tid], red, tid);
    float alpha = 1.f / (1.f + __expf(-(ga + bg[0])));
    smix[tid] = alpha * sp[tid] + (1.f - alpha) * Sl[tid];
    __syncthreads();
    {   // MLP + residual
        float acc = b1[tid];
        const float* Wp = W1 + (size_t)tid * 256;
        for (int i = 0; i < 256; ++i) acc += smix[i] * Wp[i];
        sm2[tid] = fmaxf(acc, 0.f) + smix[tid];
    }
    __syncthreads();
    float yv = block_reduce_sum(sm2[tid] * W2[tid], red, tid);
    if (tid == 0) Y[n] = yv + b2[0];
}

// (y - mean) / (std_ddof1 + 1e-8), single block
__global__ __launch_bounds__(256) void finalize_k(const float* __restrict__ Y,
                                                  float* __restrict__ out) {
    __shared__ float red[256];
    int tid = threadIdx.x;
    float s = 0.f;
    for (int i = tid; i < NN; i += 256) s += Y[i];
    float mean = block_reduce_sum(s, red, tid) * (1.0f / NN);
    float ss = 0.f;
    for (int i = tid; i < NN; i += 256) { float d = Y[i] - mean; ss += d * d; }
    float var = block_reduce_sum(ss, red, tid) * (1.0f / (NN - 1));
    float inv = 1.0f / (sqrtf(var) + 1e-8f);
    for (int i = tid; i < NN; i += 256) out[i] = (Y[i] - mean) * inv;
}

// ---------------------------------------------------------------------------
extern "C" void kernel_launch(void* const* d_in, const int* in_sizes, int n_in,
                              void* d_out, int out_size, void* d_ws, size_t ws_size,
                              hipStream_t stream) {
    const float* x      = (const float*)d_in[0];
    const float* W_ih0  = (const float*)d_in[1];
    const float* W_hh0  = (const float*)d_in[2];
    const float* b_ih0  = (const float*)d_in[3];
    const float* b_hh0  = (const float*)d_in[4];
    const float* W_ih1  = (const float*)d_in[5];
    const float* W_hh1  = (const float*)d_in[6];
    const float* b_ih1  = (const float*)d_in[7];
    const float* b_hh1  = (const float*)d_in[8];
    const float* Wq     = (const float*)d_in[9];
    const float* bq     = (const float*)d_in[10];
    const float* Wk     = (const float*)d_in[11];
    const float* bk     = (const float*)d_in[12];
    const float* Wv     = (const float*)d_in[13];
    const float* bv     = (const float*)d_in[14];
    const float* R      = (const float*)d_in[15];
    const float* W_gate = (const float*)d_in[16];
    const float* b_gate = (const float*)d_in[17];
    const float* W1     = (const float*)d_in[18];
    const float* b1     = (const float*)d_in[19];
    const float* W2     = (const float*)d_in[20];
    const float* b2     = (const float*)d_in[21];
    float* out = (float*)d_out;

    // workspace layout (bytes)
    char* ws = (char*)d_ws;
    _Float16* wih0p = (_Float16*)(ws + 0);                 // 768*64  f16
    _Float16* whh0p = (_Float16*)(ws + 98304);             // 768*256 f16
    _Float16* wih1p = (_Float16*)(ws + 491520);
    _Float16* whh1p = (_Float16*)(ws + 884736);
    float*    Sbuf  = (float*)(ws + 1277952);              // 2048*256
    float*    KH    = (float*)(ws + 3375104);              // 2048*256
    float*    VH    = (float*)(ws + 5472256);              // 2048*256
    float*    QHR   = (float*)(ws + 7569408);              // 30*256
    float*    Bbuf  = (float*)(ws + 7600128);              // 30*256
    float*    KHB   = (float*)(ws + 7630848);              // 30*256
    float*    VHB   = (float*)(ws + 7661568);              // 30*256
    float*    Ybuf  = (float*)(ws + 7692288);              // 2048

    // 1) pack weights into WMMA B-fragment layout (f16)
    {
        int tot0 = CTN * 2 * 32 * 16;   // K=64
        int tot1 = CTN * 8 * 32 * 16;   // K=256
        pack_B_k<<<(tot0 + 255) / 256, 256, 0, stream>>>(W_ih0, wih0p, INW, 2);
        pack_B_k<<<(tot1 + 255) / 256, 256, 0, stream>>>(W_hh0, whh0p, HH, 8);
        pack_B_k<<<(tot1 + 255) / 256, 256, 0, stream>>>(W_ih1, wih1p, HH, 8);
        pack_B_k<<<(tot1 + 255) / 256, 256, 0, stream>>>(W_hh1, whh1p, HH, 8);
    }

    // 2) fused 2-layer GRU scan -> Sbuf (final hidden states)
    {
        size_t smem = (size_t)(16 * G3H * 2 + G3H * 4) * 4 + (size_t)(16 * HH * 2) * 2;
        gru2_k<<<NN / 16, 256, smem, stream>>>(x, wih0p, whh0p, wih1p, whh1p,
                                               b_ih0, b_hh0, b_ih1, b_hh1, Sbuf);
    }

    // 3) projections for first MHA (keys/values from S, queries from R)
    proj_k<<<NN, 256, 0, stream>>>(Sbuf, Wk, bk, KH);
    proj_k<<<NN, 256, 0, stream>>>(Sbuf, Wv, bv, VH);
    proj_k<<<30, 256, 0, stream>>>(R, Wq, bq, QHR);

    // 4) B = mha(R, S, S)
    attnB_k<<<60, 256, 0, stream>>>(QHR, KH, VH, Bbuf);

    // 5) key/value projections of B for second MHA
    proj_k<<<30, 256, 0, stream>>>(Bbuf, Wk, bk, KHB);
    proj_k<<<30, 256, 0, stream>>>(Bbuf, Wv, bv, VHB);

    // 6) per-row: S' = mha(S,B,B), gate, MLP + residual, output head
    rowfinal_k<<<NN, 256, 0, stream>>>(Sbuf, Wq, bq, KHB, VHB,
                                       W_gate, b_gate, W1, b1, W2, b2, Ybuf);

    // 7) normalize
    finalize_k<<<1, 256, 0, stream>>>(Ybuf, out);
}
